// RTERModel_13417477833014
// MI455X (gfx1250) — matrix operand order repaired and checked by reading, not compiled, not run on hardware
//
#include <hip/hip_runtime.h>
#include <hip/hip_bf16.h>

// ---------------------------------------------------------------------------
// RTER model for MI455X (gfx1250, wave32, WMMA).
//
//  * Non-recurrent matmuls -> tiled WMMA f16 GEMM (A staged via async
//    global->LDS b128, B pre-swizzled into fragment order, streamed from L2).
//  * Recurrent GRU scans -> persistent per-(dir, batch-chunk) workgroups:
//    h in LDS, w_hh fragments preloaded in VGPRs, per-step
//    v_wmma_f32_16x16x32_f16 + fused f32 gate math.
//  * Packed-B layout: 32x16 k-tiles stored as 64 contiguous half8 chunks,
//    chunk index = c*32 + lane (c = (k&15)>>3), so a lane's v16h fragment is
//    exactly two contiguous 16B loads.
// ---------------------------------------------------------------------------

typedef __attribute__((ext_vector_type(16))) _Float16 v16h;
typedef __attribute__((ext_vector_type(8)))  _Float16 half8;
typedef __attribute__((ext_vector_type(8)))  float    v8f;

constexpr int TOK   = 65536;      // 1024 * 64 token rows
constexpr int EMBD  = 300;
constexpr int HID   = 100;
constexpr int NP3   = 320;        // 3H=300 padded
constexpr int WINDW = 40;
constexpr int NQ    = 1023;
constexpr int CTXM  = WINDW * NQ; // 40920
constexpr int LD2H  = 224;        // 2H=200 padded

// ---------------- WMMA fragment helpers (contiguous k order, same for A and B) ----------------
// A (16x32 f16): lanes 0-15 -> M rows (kbase 0), lanes 16-31 -> kbase 16.
//   element e : K = kbase+e (matches packed-B order). Two contiguous 8-half
//   runs -> compiler merges to 2x ds_load_b128 from LDS.
__device__ __forceinline__ v16h frag_a(const _Float16* P, int str, int m0,
                                       int k0, int lane) {
  int m  = m0 + (lane & 15);
  int kb = k0 + ((lane & 16) ? 16 : 0);
  v16h a;
#pragma unroll
  for (int e = 0; e < 8; ++e)  a[e] = P[m * str + kb + e];
#pragma unroll
  for (int e = 8; e < 16; ++e) a[e] = P[m * str + kb + e];
  return a;
}
// B fragment from swizzled pack: tile (kt,nt) = 64 half8 chunks; lane's
// fragment = chunks [lane] and [32+lane] -> 2 coalesced global_load_b128.
__device__ __forceinline__ v16h frag_bp(const _Float16* B, int NT, int nt,
                                        int kt, int lane) {
  const half8* p = reinterpret_cast<const half8*>(B) + (size_t)(kt * NT + nt) * 64;
  half8 lo = p[lane];
  half8 hi = p[32 + lane];
  return __builtin_shufflevector(lo, hi, 0, 1, 2, 3, 4, 5, 6, 7,
                                 8, 9, 10, 11, 12, 13, 14, 15);
}
__device__ __forceinline__ v8f wmma32(v16h a, v16h b, v8f c) {
  return __builtin_amdgcn_wmma_f32_16x16x32_f16(false, a, false, b,
                                                (short)0, c, false, false);
}
__device__ __forceinline__ float sigmoidf(float x) {
  return 1.0f / (1.0f + __expf(-x));
}

// ---------------------------------------------------------------------------
// Generic WMMA GEMM: C[M,N] = act(A[M,K]f16 * B + bias1 + bias2)
// A row-major lda (=Kpad, mult of 32). B: swizzled pack with NT n-tiles
// (NT = gridN*8). Block tile 64x128, 8 waves of 2x2 16x16 tiles.
// A tile staged in LDS via async global->LDS (full blocks) or guarded loads.
// Stores f32 (C) and/or f16 (Ch); cols >= Nact stored as 0 (zero padding).
// ---------------------------------------------------------------------------
__global__ __launch_bounds__(256) void k_gemm(
    const _Float16* __restrict__ A, int lda,
    const _Float16* __restrict__ B, int NT,
    float* __restrict__ C, _Float16* __restrict__ Ch, int ldc,
    const float* __restrict__ bias1, const float* __restrict__ bias2,
    int M, int Nact, int Nalloc, int K, int actmode)
{
  __shared__ _Float16 As[64 * 40];    // 64 rows x 32 k, stride 40
  const int tid = threadIdx.x, wave = tid >> 5, lane = tid & 31;
  const int bm = blockIdx.y, bn = blockIdx.x;
  const int wm = wave >> 2, wn = wave & 3;   // 2 x 4 wave grid
  const bool fullM = (bm * 64 + 64) <= M;    // block-uniform

  const int arow = bm * 64 + (tid >> 2);
  const int acg  = (tid & 3) * 8;
  // low 32 bits of a generic LDS pointer == LDS byte offset
  const unsigned ldsA = (unsigned)(uintptr_t)(As + (tid >> 2) * 40 + acg);

  v8f acc[2][2];
#pragma unroll
  for (int i = 0; i < 2; ++i)
#pragma unroll
    for (int j = 0; j < 2; ++j)
#pragma unroll
      for (int r = 0; r < 8; ++r) acc[i][j][r] = 0.0f;

  for (int k0 = 0; k0 < K; k0 += 32) {
    const _Float16* gp = A + (size_t)arow * lda + k0 + acg;
    if (fullM) {
      // CDNA5 async DMA: global -> LDS without touching VGPRs (ASYNCcnt).
      asm volatile("global_load_async_to_lds_b128 %0, %1, off"
                   :: "v"(ldsA), "v"(gp) : "memory");
      asm volatile("s_wait_asynccnt 0x0" ::: "memory");
    } else {
      half8 v;
      if (arow < M) {
        v = *reinterpret_cast<const half8*>(gp);
      } else {
#pragma unroll
        for (int e = 0; e < 8; ++e) v[e] = (_Float16)0.0f;
      }
      *reinterpret_cast<half8*>(As + (tid >> 2) * 40 + acg) = v;
    }
    __syncthreads();
    const int kt = k0 >> 5;
    v16h af[2], bf[2];
#pragma unroll
    for (int i = 0; i < 2; ++i) af[i] = frag_a(As, 40, (wm * 2 + i) * 16, 0, lane);
#pragma unroll
    for (int j = 0; j < 2; ++j) bf[j] = frag_bp(B, NT, bn * 8 + wn * 2 + j, kt, lane);
#pragma unroll
    for (int i = 0; i < 2; ++i)
#pragma unroll
      for (int j = 0; j < 2; ++j) acc[i][j] = wmma32(af[i], bf[j], acc[i][j]);
    __syncthreads();
  }

  const int mb = (lane & 16) ? 8 : 0;
  const int nl = lane & 15;
#pragma unroll
  for (int i = 0; i < 2; ++i)
#pragma unroll
    for (int j = 0; j < 2; ++j) {
      int col = bn * 128 + wn * 32 + j * 16 + nl;
      if (col >= Nalloc) continue;
      bool valid = col < Nact;
      float bv = 0.0f;
      if (valid) {
        if (bias1) bv += bias1[col];
        if (bias2) bv += bias2[col];
      }
#pragma unroll
      for (int r = 0; r < 8; ++r) {
        int row = bm * 64 + wm * 32 + i * 16 + mb + r;
        if (row < M) {
          float v = 0.0f;
          if (valid) {
            v = acc[i][j][r] + bv;
            if (actmode == 1) v = tanhf(v);
          }
          if (C)  C[(size_t)row * ldc + col] = v;
          if (Ch) Ch[(size_t)row * ldc + col] = (_Float16)v;
        }
      }
    }
}

// ---------------------------------------------------------------------------
// Persistent GRU scan. grid = (ceil(Btot/32), 2 dirs). Per step:
// gh = h @ w_hh^T via WMMA (w_hh fragments preloaded in VGPRs), then fused
// gates: r=sig(ir+hr) z=sig(iz+hz) n=tanh(in+r*hn) h=(1-z)n+z h.
// xg: f16 [T*Btot x 320] (b_ih folded by GEMM). out: f16 [T*Btot x ldo],
// fwd at col 0, bwd at col 100. whh pack: NT=20 swizzled tiles.
// ---------------------------------------------------------------------------
__global__ __launch_bounds__(256) void k_gru_scan(
    const _Float16* __restrict__ xg_f, const _Float16* __restrict__ xg_b,
    const _Float16* __restrict__ whh_f, const _Float16* __restrict__ whh_b,
    const float* __restrict__ bhh_f, const float* __restrict__ bhh_b,
    _Float16* __restrict__ out, int T, int Btot, int ldo)
{
  __shared__ float    Gh[32 * 324];   // 32 rows x 3H(320) padded stride 324
  __shared__ _Float16 Hs[32 * 136];   // h state, K padded 128, stride 136
  const int tid = threadIdx.x, wave = tid >> 5, lane = tid & 31;
  const int dir = blockIdx.y;
  const _Float16* xg  = dir ? xg_b : xg_f;
  const _Float16* W   = dir ? whh_b : whh_f;
  const float*    bhh = dir ? bhh_b : bhh_f;
  const int colOff = dir ? HID : 0;
  const int b0 = blockIdx.x * 32;

  for (int i = tid; i < 32 * 136; i += 256) Hs[i] = (_Float16)0.0f;

  // Preload w_hh fragments: 5 tiles x 4 k-steps = 160 VGPRs/lane.
  v16h wf[5][4];
#pragma unroll
  for (int c = 0; c < 5; ++c) {
    int tau = wave + 8 * c;            // tile id in [0,40): mt=tau&1, nt=tau>>1
#pragma unroll
    for (int kt = 0; kt < 4; ++kt)
      wf[c][kt] = frag_bp(W, 20, tau >> 1, kt, lane);
  }
  __syncthreads();

  for (int s = 0; s < T; ++s) {
    const int t = dir ? (T - 1 - s) : s;
#pragma unroll
    for (int c = 0; c < 5; ++c) {
      const int tau = wave + 8 * c;
      const int mt = tau & 1, nt = tau >> 1;
      v8f acc;
#pragma unroll
      for (int r = 0; r < 8; ++r) acc[r] = 0.0f;
#pragma unroll
      for (int kt = 0; kt < 4; ++kt) {
        v16h a = frag_a(Hs, 136, mt * 16, kt * 32, lane);
        acc = wmma32(a, wf[c][kt], acc);
      }
      const int n  = nt * 16 + (lane & 15);
      const int mb = (lane & 16) ? 8 : 0;
#pragma unroll
      for (int r = 0; r < 8; ++r) Gh[(mt * 16 + mb + r) * 324 + n] = acc[r];
    }
    __syncthreads();
    for (int e = tid; e < 32 * HID; e += 256) {
      const int b = e / HID, j = e - b * HID;
      const int gb = b0 + b;
      if (gb < Btot) {
        const size_t row = (size_t)t * Btot + gb;
        float ir = (float)xg[row * NP3 + j];
        float iz = (float)xg[row * NP3 + j + HID];
        float in = (float)xg[row * NP3 + j + 2 * HID];
        float hr = Gh[b * 324 + j]           + bhh[j];
        float hz = Gh[b * 324 + j + HID]     + bhh[j + HID];
        float hn = Gh[b * 324 + j + 2 * HID] + bhh[j + 2 * HID];
        float r  = sigmoidf(ir + hr);
        float z  = sigmoidf(iz + hz);
        float nn = tanhf(in + r * hn);
        float hp = (float)Hs[b * 136 + j];
        float hv = (1.0f - z) * nn + z * hp;
        Hs[b * 136 + j] = (_Float16)hv;
        out[row * ldo + colOff + j] = (_Float16)hv;
      }
    }
    __syncthreads();
  }
}

// ---------------------------------------------------------------------------
// Attention GRU scan (one direction). X f32 [T*Btot x 256]: Xr at cols 0..99,
// Xw at cols 128..227. U pack = [Ur | U], NT=16 swizzled tiles. Per step:
// r = sig(Xr + hUr); ht = tanh(Xw + r*(hU + bU)); h = g*ht + (1-g)*h.
// Only final h is written out.
// ---------------------------------------------------------------------------
__global__ __launch_bounds__(256) void k_attn_scan(
    const float* __restrict__ X, const _Float16* __restrict__ U,
    const float* __restrict__ bu, const float* __restrict__ sc,
    float* __restrict__ hout, int T, int Btot, int rev)
{
  __shared__ float    Gh[32 * 260];
  __shared__ _Float16 Hs[32 * 136];
  const int tid = threadIdx.x, wave = tid >> 5, lane = tid & 31;
  const int b0 = blockIdx.x * 32;

  for (int i = tid; i < 32 * 136; i += 256) Hs[i] = (_Float16)0.0f;

  v16h wf[4][4];                       // 32 tiles / 8 waves = 4 each
#pragma unroll
  for (int c = 0; c < 4; ++c) {
    int tau = wave + 8 * c;            // mt=tau&1, nt=tau>>1 in [0,16)
#pragma unroll
    for (int kt = 0; kt < 4; ++kt)
      wf[c][kt] = frag_bp(U, 16, tau >> 1, kt, lane);
  }
  __syncthreads();

  for (int s = 0; s < T; ++s) {
    const int t = rev ? (T - 1 - s) : s;
#pragma unroll
    for (int c = 0; c < 4; ++c) {
      const int tau = wave + 8 * c;
      const int mt = tau & 1, nt = tau >> 1;
      v8f acc;
#pragma unroll
      for (int r = 0; r < 8; ++r) acc[r] = 0.0f;
#pragma unroll
      for (int kt = 0; kt < 4; ++kt) {
        v16h a = frag_a(Hs, 136, mt * 16, kt * 32, lane);
        acc = wmma32(a, wf[c][kt], acc);
      }
      const int n  = nt * 16 + (lane & 15);
      const int mb = (lane & 16) ? 8 : 0;
#pragma unroll
      for (int r = 0; r < 8; ++r) Gh[(mt * 16 + mb + r) * 260 + n] = acc[r];
    }
    __syncthreads();
    for (int e = tid; e < 32 * HID; e += 256) {
      const int b = e / HID, j = e - b * HID;
      const int gb = b0 + b;
      if (gb < Btot) {
        const size_t row = (size_t)t * Btot + gb;
        float r  = sigmoidf(X[row * 256 + j] + Gh[b * 260 + j]);
        float ht = tanhf(X[row * 256 + 128 + j] +
                         r * (Gh[b * 260 + 128 + j] + bu[j]));
        float g  = sc[(size_t)gb * WINDW + t];
        float hp = (float)Hs[b * 136 + j];
        float hv = g * ht + (1.0f - g) * hp;
        Hs[b * 136 + j] = (_Float16)hv;
      }
    }
    __syncthreads();
  }
  for (int e = tid; e < 32 * HID; e += 256) {
    const int b = e / HID, j = e - b * HID;
    const int gb = b0 + b;
    if (gb < Btot) hout[(size_t)gb * HID + j] = (float)Hs[b * 136 + j];
  }
}

// ------------------------------- small kernels ------------------------------
__global__ void k_gather_embed(const int* __restrict__ ids,
                               const float* __restrict__ emb,
                               _Float16* __restrict__ act) {
  int i = blockIdx.x * 256 + threadIdx.x;
  if (i >= TOK * NP3) return;
  int r = i / NP3, c = i - r * NP3;
  float v = 0.0f;
  if (c < EMBD) v = emb[(size_t)ids[r] * EMBD + c];
  act[i] = (_Float16)v;
}

// pack f32 src[N][K] -> f16 swizzled B tiles (NT n-tiles), col offset n0.
// Tile (kt,nt): 64 half8 chunks; half offset within tile:
//   (( (k&15)>>3 )*32 + lane)*8 + (k&7), lane = ((k&16)) | (n&15).
__global__ void k_pack_b(const float* __restrict__ src, _Float16* __restrict__ dst,
                         int N, int K, int Kpad, int NT, int n0) {
  int i = blockIdx.x * 256 + threadIdx.x;
  if (i >= Kpad * N) return;
  int k = i / N, nr = i - k * N;
  int n = n0 + nr;
  int kt = k >> 5, kin = k & 31;
  int nt = n >> 4, nin = n & 15;
  int lane = ((kin >> 4) << 4) | nin;
  int kl = kin & 15;
  size_t off = ((size_t)(kt * NT + nt) * 64 + (size_t)((kl >> 3) * 32 + lane)) * 8
               + (kl & 7);
  dst[off] = (_Float16)((k < K) ? src[(size_t)nr * K + k] : 0.0f);
}

__global__ void k_bias_att(const float* __restrict__ br, const float* __restrict__ bur,
                           const float* __restrict__ bw, float* __restrict__ dst) {
  int j = blockIdx.x * 256 + threadIdx.x;
  if (j >= 256) return;
  float v = 0.0f;
  if (j < HID) v = br[j] + bur[j];
  else if (j >= 128 && j < 128 + HID) v = bw[j - 128];
  dst[j] = v;
}

__global__ void k_maxpool(const _Float16* __restrict__ h, _Float16* __restrict__ mp) {
  int i = blockIdx.x * 256 + threadIdx.x;
  if (i >= 1024 * LD2H) return;
  int t = i / LD2H, c = i - t * LD2H;
  float m = -3.4e38f;
  for (int b = 0; b < 64; ++b)
    m = fmaxf(m, (float)h[((size_t)t * 64 + b) * LD2H + c]);
  mp[i] = (_Float16)m;
}

__global__ void k_build_windows(const _Float16* __restrict__ u16,
                                _Float16* __restrict__ win) {
  int i = blockIdx.x * 256 + threadIdx.x;
  if (i >= CTXM * 128) return;
  int row = i >> 7, c = i & 127;
  int w = row / NQ, q = row - w * NQ;
  int src = q + 1 - WINDW + w;
  _Float16 v = (_Float16)0.0f;
  if (src >= 0 && c < HID) v = u16[(size_t)src * 128 + c];
  win[i] = v;
}

__global__ void k_mem(const _Float16* __restrict__ win, const _Float16* __restrict__ octx,
                      float* __restrict__ memb, _Float16* __restrict__ inp) {
  int i = blockIdx.x * 256 + threadIdx.x;
  if (i >= NQ * WINDW * HID) return;
  int j = i % HID; int qw = i / HID; int w = qw % WINDW; int q = qw / WINDW;
  size_t rin = (size_t)w * NQ + q;
  float v = (float)win[rin * 128 + j] + (float)octx[rin * LD2H + j] +
            (float)octx[rin * LD2H + HID + j];
  memb[((size_t)q * WINDW + w) * HID + j] = v;
  inp[rin * 128 + j] = (_Float16)v;
}

__global__ void k_eps_init(const float* __restrict__ u32, float* __restrict__ eps) {
  int i = blockIdx.x * 256 + threadIdx.x;
  if (i >= NQ * 128) return;
  int q = i >> 7, j = i & 127;
  eps[i] = u32[(size_t)(q + 1) * 128 + j];
}

__global__ void k_attn_scores(const float* __restrict__ eps,
                              const float* __restrict__ memb,
                              float* __restrict__ sc) {
  __shared__ float s[WINDW];
  __shared__ float red[2];
  int q = blockIdx.x, tx = threadIdx.x;
  if (tx < WINDW) {
    int idx = q + 1 - WINDW + tx;
    float a = -1.0e10f;
    if (idx >= 0) {
      float acc = 0.0f;
      for (int j = 0; j < HID; ++j)
        acc += eps[(size_t)q * 128 + j] * memb[((size_t)q * WINDW + tx) * HID + j];
      a = acc;
    }
    s[tx] = a;
  }
  __syncthreads();
  if (tx == 0) {
    float m = -3.4e38f;
    for (int w = 0; w < WINDW; ++w) m = fmaxf(m, s[w]);
    float ss = 0.0f;
    for (int w = 0; w < WINDW; ++w) ss += __expf(s[w] - m);
    red[0] = m; red[1] = ss;
  }
  __syncthreads();
  if (tx < WINDW) sc[(size_t)q * WINDW + tx] = __expf(s[tx] - red[0]) / red[1];
}

__global__ void k_eps_update(float* __restrict__ eps, const float* __restrict__ hf,
                             const float* __restrict__ hb) {
  int i = blockIdx.x * 256 + threadIdx.x;
  if (i >= NQ * HID) return;
  int q = i / HID, j = i - q * HID;
  eps[(size_t)q * 128 + j] += hf[i] + hb[i];
}

__global__ void k_classifier(const float* __restrict__ eps, const float* __restrict__ w,
                             const float* __restrict__ b, float* __restrict__ out) {
  int i = blockIdx.x * 256 + threadIdx.x;
  if (i >= NQ * 7) return;
  int q = i / 7, c = i - q * 7;
  float acc = b[c];
  for (int j = 0; j < HID; ++j) acc += eps[(size_t)q * 128 + j] * w[(size_t)c * HID + j];
  out[i] = acc;
}

// ---------------------------------------------------------------------------
extern "C" void kernel_launch(void* const* d_in, const int* in_sizes, int n_in,
                              void* d_out, int out_size, void* d_ws, size_t ws_size,
                              hipStream_t stream)
{
  (void)in_sizes; (void)n_in; (void)out_size; (void)ws_size;
  // Input order: JAX-pytree flatten (sorted dict keys, lists in order):
  // dialogue_ids, params{ att[3]{bwd,fwd}{U,Ur,W,Wr}{b,w}, classifier{b,w},
  // ctx_layers[1]{b,f}{b_hh,b_ih,w_hh,w_ih}, emb, linear{b,w},
  // utt_layers[2]{b,f}{b_hh,b_ih,w_hh,w_ih} }  -> 78 arrays total.
  int p = 0;
  const int* d_ids = (const int*)d_in[p++];
  const float *aUb[3][2], *aUw[3][2], *aUrb[3][2], *aUrw[3][2];
  const float *aWb[3][2], *aWw[3][2], *aWrb[3][2], *aWrw[3][2];
  for (int h = 0; h < 3; ++h)
    for (int d = 0; d < 2; ++d) {          // d: 0=bwd, 1=fwd
      aUb[h][d]  = (const float*)d_in[p++];
      aUw[h][d]  = (const float*)d_in[p++];
      aUrb[h][d] = (const float*)d_in[p++];
      aUrw[h][d] = (const float*)d_in[p++];
      aWb[h][d]  = (const float*)d_in[p++];
      aWw[h][d]  = (const float*)d_in[p++];
      aWrb[h][d] = (const float*)d_in[p++];
      aWrw[h][d] = (const float*)d_in[p++];
    }
  const float* cls_b = (const float*)d_in[p++];
  const float* cls_w = (const float*)d_in[p++];
  const float *c_bhh[2], *c_bih[2], *c_whh[2], *c_wih[2];
  for (int d = 0; d < 2; ++d) {
    c_bhh[d] = (const float*)d_in[p++]; c_bih[d] = (const float*)d_in[p++];
    c_whh[d] = (const float*)d_in[p++]; c_wih[d] = (const float*)d_in[p++];
  }
  const float* embp  = (const float*)d_in[p++];
  const float* lin_b = (const float*)d_in[p++];
  const float* lin_w = (const float*)d_in[p++];
  const float *u_bhh[2][2], *u_bih[2][2], *u_whh[2][2], *u_wih[2][2];
  for (int l = 0; l < 2; ++l)
    for (int d = 0; d < 2; ++d) {
      u_bhh[l][d] = (const float*)d_in[p++]; u_bih[l][d] = (const float*)d_in[p++];
      u_whh[l][d] = (const float*)d_in[p++]; u_wih[l][d] = (const float*)d_in[p++];
    }

  // ---- workspace layout (bump allocator, ~147 MB total) ----
  char* base = (char*)d_ws; size_t off = 0;
  auto alloc = [&](size_t bytes) -> void* {
    void* ptr = base + off; off = (off + bytes + 255) & ~(size_t)255; return ptr;
  };
  const size_t SZ_BIG = (size_t)TOK * NP3 * 2;            // 41.94 MB
  _Float16* bufA = (_Float16*)alloc(SZ_BIG);              // act0 / h / win+inp+octx
  _Float16* xgF  = (_Float16*)alloc(SZ_BIG);              // xg fwd / X fwd (f32)
  _Float16* xgB  = (_Float16*)alloc(SZ_BIG);              // xg bwd / X bwd (f32)
  _Float16* maxp = (_Float16*)alloc((size_t)1024 * LD2H * 2);
  float* u32  = (float*)alloc((size_t)1024 * 128 * 4);
  _Float16* u16p = (_Float16*)alloc((size_t)1024 * 128 * 2);
  float* memb = (float*)alloc((size_t)NQ * WINDW * HID * 4);
  float* eps  = (float*)alloc((size_t)NQ * 128 * 4);
  float* sc   = (float*)alloc((size_t)NQ * WINDW * 4);
  float* hf   = (float*)alloc((size_t)NQ * HID * 4);
  float* hb   = (float*)alloc((size_t)NQ * HID * 4);
  // swizzled weight packs: bytes = kt_count * NT * 512 halves * 2
  _Float16* pWih0[2]; for (int d = 0; d < 2; ++d) pWih0[d] = (_Float16*)alloc((size_t)10 * 24 * 512 * 2);
  _Float16* pWih1[2]; for (int d = 0; d < 2; ++d) pWih1[d] = (_Float16*)alloc((size_t)7 * 24 * 512 * 2);
  _Float16* pWhh[2][2];
  for (int l = 0; l < 2; ++l)
    for (int d = 0; d < 2; ++d) pWhh[l][d] = (_Float16*)alloc((size_t)4 * 20 * 512 * 2);
  _Float16* pCwih[2]; for (int d = 0; d < 2; ++d) pCwih[d] = (_Float16*)alloc((size_t)4 * 24 * 512 * 2);
  _Float16* pCwhh[2]; for (int d = 0; d < 2; ++d) pCwhh[d] = (_Float16*)alloc((size_t)4 * 20 * 512 * 2);
  _Float16* pLin = (_Float16*)alloc((size_t)7 * 8 * 512 * 2);
  _Float16* pAttW[3][2]; _Float16* pAttU[3][2]; float* pAttBias[3][2];
  for (int h = 0; h < 3; ++h)
    for (int d = 0; d < 2; ++d) {
      pAttW[h][d] = (_Float16*)alloc((size_t)4 * 16 * 512 * 2);
      pAttU[h][d] = (_Float16*)alloc((size_t)4 * 16 * 512 * 2);
      pAttBias[h][d] = (float*)alloc(256 * 4);
    }
  // aliases inside bufA (live ranges do not overlap)
  _Float16* hbuf = bufA;
  _Float16* win  = bufA;
  _Float16* inp  = (_Float16*)((char*)bufA + (size_t)CTXM * 128 * 2);
  _Float16* octx = (_Float16*)((char*)bufA + (size_t)CTXM * 128 * 2 * 2);

  auto pack = [&](const float* s, _Float16* dst, int N, int K, int Kpad, int NT, int n0) {
    int tot = Kpad * N;
    k_pack_b<<<(tot + 255) / 256, 256, 0, stream>>>(s, dst, N, K, Kpad, NT, n0);
  };
  auto gemm = [&](const _Float16* A, int lda, const _Float16* B, int NT,
                  float* C, _Float16* Ch, int ldc, const float* b1, const float* b2,
                  int M, int Nact, int Nalloc, int K, int actm) {
    dim3 g((Nalloc + 127) / 128, (M + 63) / 64);
    k_gemm<<<g, 256, 0, stream>>>(A, lda, B, NT, C, Ch, ldc, b1, b2, M, Nact, Nalloc, K, actm);
  };

  // ---- weight packing ----
  for (int d = 0; d < 2; ++d) {
    pack(u_wih[0][d], pWih0[d], 300, 300, 320, 24, 0);
    pack(u_wih[1][d], pWih1[d], 300, 200, 224, 24, 0);
    pack(u_whh[0][d], pWhh[0][d], 300, 100, 128, 20, 0);
    pack(u_whh[1][d], pWhh[1][d], 300, 100, 128, 20, 0);
    pack(c_wih[d], pCwih[d], 300, 100, 128, 24, 0);
    pack(c_whh[d], pCwhh[d], 300, 100, 128, 20, 0);
  }
  pack(lin_w, pLin, 100, 200, 224, 8, 0);
  for (int h = 0; h < 3; ++h)
    for (int d = 0; d < 2; ++d) {
      pack(aWrw[h][d], pAttW[h][d], 100, 100, 128, 16, 0);     // Wr -> cols 0..
      pack(aWw[h][d],  pAttW[h][d], 100, 100, 128, 16, 128);   // W  -> cols 128..
      pack(aUrw[h][d], pAttU[h][d], 100, 100, 128, 16, 0);     // Ur
      pack(aUw[h][d],  pAttU[h][d], 100, 100, 128, 16, 128);   // U
      k_bias_att<<<1, 256, 0, stream>>>(aWrb[h][d], aUrb[h][d], aWb[h][d], pAttBias[h][d]);
    }

  // ---- utterance pipeline ----
  k_gather_embed<<<(TOK * NP3 + 255) / 256, 256, 0, stream>>>(d_ids, embp, bufA);
  gemm(bufA, NP3, pWih0[1], 24, nullptr, xgF, NP3, u_bih[0][1], nullptr, TOK, 300, NP3, 320, 0);
  gemm(bufA, NP3, pWih0[0], 24, nullptr, xgB, NP3, u_bih[0][0], nullptr, TOK, 300, NP3, 320, 0);
  hipMemsetAsync(hbuf, 0, (size_t)TOK * LD2H * 2, stream);
  k_gru_scan<<<dim3(2, 2), 256, 0, stream>>>(xgF, xgB, pWhh[0][1], pWhh[0][0],
                                             u_bhh[0][1], u_bhh[0][0], hbuf, 1024, 64, LD2H);
  gemm(hbuf, LD2H, pWih1[1], 24, nullptr, xgF, NP3, u_bih[1][1], nullptr, TOK, 300, NP3, 224, 0);
  gemm(hbuf, LD2H, pWih1[0], 24, nullptr, xgB, NP3, u_bih[1][0], nullptr, TOK, 300, NP3, 224, 0);
  hipMemsetAsync(hbuf, 0, (size_t)TOK * LD2H * 2, stream);
  k_gru_scan<<<dim3(2, 2), 256, 0, stream>>>(xgF, xgB, pWhh[1][1], pWhh[1][0],
                                             u_bhh[1][1], u_bhh[1][0], hbuf, 1024, 64, LD2H);
  k_maxpool<<<(1024 * LD2H + 255) / 256, 256, 0, stream>>>(hbuf, maxp);
  gemm(maxp, LD2H, pLin, 8, u32, u16p, 128, lin_b, nullptr, 1024, 100, 128, 224, 1);

  // ---- context pipeline ----
  k_build_windows<<<(CTXM * 128 + 255) / 256, 256, 0, stream>>>(u16p, win);
  gemm(win, 128, pCwih[1], 24, nullptr, xgF, NP3, c_bih[1], nullptr, CTXM, 300, NP3, 128, 0);
  gemm(win, 128, pCwih[0], 24, nullptr, xgB, NP3, c_bih[0], nullptr, CTXM, 300, NP3, 128, 0);
  hipMemsetAsync(octx, 0, (size_t)CTXM * LD2H * 2, stream);
  k_gru_scan<<<dim3(32, 2), 256, 0, stream>>>(xgF, xgB, pCwhh[1], pCwhh[0],
                                              c_bhh[1], c_bhh[0], octx, WINDW, NQ, LD2H);
  hipMemsetAsync(inp, 0, (size_t)CTXM * 128 * 2, stream);
  k_mem<<<(NQ * WINDW * HID + 255) / 256, 256, 0, stream>>>(win, octx, memb, inp);
  k_eps_init<<<(NQ * 128 + 255) / 256, 256, 0, stream>>>(u32, eps);

  // ---- attention hops ----
  for (int h = 0; h < 3; ++h) {
    k_attn_scores<<<NQ, 64, 0, stream>>>(eps, memb, sc);
    float* Xf = (float*)xgF; float* Xb = (float*)xgB;
    gemm(inp, 128, pAttW[h][1], 16, Xf, nullptr, 256, pAttBias[h][1], nullptr, CTXM, 256, 256, 128, 0);
    gemm(inp, 128, pAttW[h][0], 16, Xb, nullptr, 256, pAttBias[h][0], nullptr, CTXM, 256, 256, 128, 0);
    k_attn_scan<<<dim3(32), 256, 0, stream>>>(Xf, pAttU[h][1], aUb[h][1], sc, hf, WINDW, NQ, 0);
    k_attn_scan<<<dim3(32), 256, 0, stream>>>(Xb, pAttU[h][0], aUb[h][0], sc, hb, WINDW, NQ, 1);
    k_eps_update<<<(NQ * HID + 255) / 256, 256, 0, stream>>>(eps, hf, hb);
  }
  k_classifier<<<(NQ * 7 + 255) / 256, 256, 0, stream>>>(eps, cls_w, cls_b, (float*)d_out);
}
